// Net_59476707115199
// MI455X (gfx1250) — matrix operand order repaired and checked by reading, not compiled, not run on hardware
//
#include <hip/hip_runtime.h>
#include <hip/hip_bf16.h>

typedef __attribute__((ext_vector_type(16))) _Float16 v16h;
typedef __attribute__((ext_vector_type(8)))  _Float16 v8h;
typedef __attribute__((ext_vector_type(8)))  float    v8f;

#define NNODES 50000
#define KIN    133
#define KPAD   160     // 5 K-tiles of 32 (zeros in [133,160))
#define XSTR   168     // LDS row stride in halves: 336B = 84 dwords, conflict-free b128
#define D1     64      // H=8, C=8
#define NSLOPE 0.2f

__device__ __forceinline__ float leaky(float x) { return x > 0.0f ? x : NSLOPE * x; }

__device__ __forceinline__ v16h cat16(v8h lo, v8h hi) {
    return __builtin_shufflevector(lo, hi, 0,1,2,3,4,5,6,7,8,9,10,11,12,13,14,15);
}

// Correct float atomic-max for mixed-sign values (no NaNs in this workload):
// nonneg floats ordered as int bits, negative floats reverse-ordered as uint bits.
__device__ __forceinline__ void atomicMaxF(float* addr, float v) {
    int iv = __float_as_int(v);
    if (iv >= 0) atomicMax((int*)addr, iv);
    else         atomicMin((unsigned int*)addr, (unsigned int)iv);
}

// ---------------------------------------------------------------------------
// Kernel 1: h1 = x @ W1 via v_wmma_f32_16x16x32_f16.
// Block = 256 threads = 8 waves; wave w owns rows [blk*128 + w*16, +16), all 64 cols.
// W staged TRANSPOSED in LDS so B fragments are contiguous-K -> ds_load_b128.
// ---------------------------------------------------------------------------
__global__ __launch_bounds__(256) void gemm1_wmma(const float* __restrict__ x,
                                                  const float* __restrict__ W1,
                                                  float* __restrict__ h1) {
    __shared__ __align__(16) _Float16 Wt[D1 * XSTR];       // [n][k], 21504 B
    __shared__ __align__(16) _Float16 Xl[8][16 * XSTR];    // per-wave A tiles, 43008 B

    const int tid  = threadIdx.x;
    const int wave = tid >> 5;
    const int lane = tid & 31;

    // Stage W1 transposed (fp32 -> fp16), zero-pad K in [133,168)
    for (int i = tid; i < D1 * XSTR; i += 256) {
        int n = i / XSTR, k = i - n * XSTR;
        Wt[i] = (k < KIN) ? (_Float16)W1[k * D1 + n] : (_Float16)0.0f;
    }

    const int rowBase = blockIdx.x * 128 + wave * 16;
    if (rowBase < NNODES) {  // NNODES % 16 == 0 -> tiles are all-or-nothing
        _Float16* xs = &Xl[wave][0];
        // 16 rows x 133 floats are contiguous and 16B-aligned: b128 global loads
        const float4* xv = (const float4*)(x + (size_t)rowBase * KIN);
        for (int i = lane; i < (16 * KIN) / 4; i += 32) {   // 532 float4s
            float4 f = xv[i];
            float vals[4] = {f.x, f.y, f.z, f.w};
            #pragma unroll
            for (int c = 0; c < 4; ++c) {
                int idx = i * 4 + c;
                int r = idx / KIN, k = idx - r * KIN;
                xs[r * XSTR + k] = (_Float16)vals[c];
            }
        }
        // zero-fill pad columns [133,168)
        for (int i = lane; i < 16 * (XSTR - KIN); i += 32) {
            int r = i / (XSTR - KIN), k = KIN + (i - r * (XSTR - KIN));
            xs[r * XSTR + k] = (_Float16)0.0f;
        }
    }
    __syncthreads();
    if (rowBase >= NNODES) return;

    const int m     = lane & 15;
    const int abase = (lane < 16) ? 0 : 16;   // A: K sub-block per lane half
    const int bbase = (lane < 16) ? 0 : 16;   // B: K sub-block per lane half
    const _Float16* xrow = &Xl[wave][m * XSTR];

    v8f acc0 = {}, acc1 = {}, acc2 = {}, acc3 = {};
    for (int kt = 0; kt < 5; ++kt) {
        const int k0 = kt * 32;
        // A fragment (16x32 f16): elem e -> K=abase+e (contiguous)
        v16h a = cat16(*(const v8h*)(xrow + k0 + abase),
                       *(const v8h*)(xrow + k0 + abase + 8));
        // B fragments (32x16 f16): elem e -> K = bbase+e (contiguous in transposed Wt)
        const _Float16* w0 = &Wt[(m +  0) * XSTR + k0 + bbase];
        const _Float16* w1 = &Wt[(m + 16) * XSTR + k0 + bbase];
        const _Float16* w2 = &Wt[(m + 32) * XSTR + k0 + bbase];
        const _Float16* w3 = &Wt[(m + 48) * XSTR + k0 + bbase];
        v16h b0 = cat16(*(const v8h*)(w0), *(const v8h*)(w0 + 8));
        v16h b1 = cat16(*(const v8h*)(w1), *(const v8h*)(w1 + 8));
        v16h b2 = cat16(*(const v8h*)(w2), *(const v8h*)(w2 + 8));
        v16h b3 = cat16(*(const v8h*)(w3), *(const v8h*)(w3 + 8));
        acc0 = __builtin_amdgcn_wmma_f32_16x16x32_f16(false, a, false, b0, (short)0, acc0, false, false);
        acc1 = __builtin_amdgcn_wmma_f32_16x16x32_f16(false, a, false, b1, (short)0, acc1, false, false);
        acc2 = __builtin_amdgcn_wmma_f32_16x16x32_f16(false, a, false, b2, (short)0, acc2, false, false);
        acc3 = __builtin_amdgcn_wmma_f32_16x16x32_f16(false, a, false, b3, (short)0, acc3, false, false);
    }

    // D layout: VGPR v -> M = v (lanes 0-15) / 8+v (lanes 16-31); N = lane&15
    const int mrow = rowBase + ((lane < 16) ? 0 : 8);
    #pragma unroll
    for (int v = 0; v < 8; ++v) {
        int r = (mrow + v) * D1;
        h1[r + m +  0] = acc0[v];
        h1[r + m + 16] = acc1[v];
        h1[r + m + 32] = acc2[v];
        h1[r + m + 48] = acc3[v];
    }
}

// ---------------------------------------------------------------------------
// Layer-1 softmax / aggregation passes (H = 8)
// ---------------------------------------------------------------------------
__global__ void node1_att(const float* __restrict__ h1,
                          const float* __restrict__ att_s, const float* __restrict__ att_d,
                          float* __restrict__ as1, float* __restrict__ ad1,
                          float* __restrict__ emax1) {
    int t = blockIdx.x * blockDim.x + threadIdx.x;      // t = n*8 + h
    if (t >= NNODES * 8) return;
    int n = t >> 3, h = t & 7;
    float s = 0.f, d = 0.f;
    #pragma unroll
    for (int c = 0; c < 8; ++c) {
        float v = h1[n * D1 + h * 8 + c];
        s += v * att_s[h * 8 + c];
        d += v * att_d[h * 8 + c];
    }
    as1[t] = s; ad1[t] = d;
    emax1[t] = leaky(s + d);        // self-loop logit seeds the running max
}

__global__ void edge1_max(const int* __restrict__ src, const int* __restrict__ dst, int E,
                          const float* __restrict__ as1, const float* __restrict__ ad1,
                          float* __restrict__ emax1) {
    int t = blockIdx.x * blockDim.x + threadIdx.x;
    if (t >= E * 8) return;
    int e = t >> 3, h = t & 7;
    int s = src[e], d = dst[e];
    atomicMaxF(&emax1[d * 8 + h], leaky(as1[s * 8 + h] + ad1[d * 8 + h]));
}

__global__ void node1_denom_init(const float* __restrict__ as1, const float* __restrict__ ad1,
                                 const float* __restrict__ emax1, float* __restrict__ denom1) {
    int t = blockIdx.x * blockDim.x + threadIdx.x;
    if (t >= NNODES * 8) return;
    denom1[t] = __expf(leaky(as1[t] + ad1[t]) - emax1[t]);   // self-loop term
}

__global__ void edge1_denom(const int* __restrict__ src, const int* __restrict__ dst, int E,
                            const float* __restrict__ as1, const float* __restrict__ ad1,
                            const float* __restrict__ emax1, float* __restrict__ denom1) {
    int t = blockIdx.x * blockDim.x + threadIdx.x;
    if (t >= E * 8) return;
    int e = t >> 3, h = t & 7;
    int s = src[e], d = dst[e];
    float ee = leaky(as1[s * 8 + h] + ad1[d * 8 + h]);
    atomicAdd(&denom1[d * 8 + h], __expf(ee - emax1[d * 8 + h]));
}

__global__ void node1_self_msg(const float* __restrict__ h1,
                               const float* __restrict__ as1, const float* __restrict__ ad1,
                               const float* __restrict__ emax1, const float* __restrict__ denom1,
                               float* __restrict__ out1) {
    int t = blockIdx.x * blockDim.x + threadIdx.x;      // t = n*64 + h*8 + c
    if (t >= NNODES * D1) return;
    int n = t >> 6, h = (t >> 3) & 7;
    float es = leaky(as1[n * 8 + h] + ad1[n * 8 + h]);
    float alpha = __expf(es - emax1[n * 8 + h]) / fmaxf(denom1[n * 8 + h], 1e-16f);
    out1[t] = alpha * h1[t];                            // initializes out1
}

__global__ void edge1_msg(const int* __restrict__ src, const int* __restrict__ dst, int E,
                          const float* __restrict__ h1,
                          const float* __restrict__ as1, const float* __restrict__ ad1,
                          const float* __restrict__ emax1, const float* __restrict__ denom1,
                          float* __restrict__ out1) {
    int t = blockIdx.x * blockDim.x + threadIdx.x;
    if (t >= E * 8) return;
    int e = t >> 3, h = t & 7;
    int s = src[e], d = dst[e];
    float ee = leaky(as1[s * 8 + h] + ad1[d * 8 + h]);
    float alpha = __expf(ee - emax1[d * 8 + h]) / fmaxf(denom1[d * 8 + h], 1e-16f);
    const float4* hs = (const float4*)&h1[s * D1 + h * 8];
    float4 v0 = hs[0], v1 = hs[1];
    float* o = &out1[d * D1 + h * 8];
    atomicAdd(o + 0, alpha * v0.x); atomicAdd(o + 1, alpha * v0.y);
    atomicAdd(o + 2, alpha * v0.z); atomicAdd(o + 3, alpha * v0.w);
    atomicAdd(o + 4, alpha * v1.x); atomicAdd(o + 5, alpha * v1.y);
    atomicAdd(o + 6, alpha * v1.z); atomicAdd(o + 7, alpha * v1.w);
}

__global__ void node1_bias_elu(const float* __restrict__ b1, float* __restrict__ out1) {
    int t = blockIdx.x * blockDim.x + threadIdx.x;
    if (t >= NNODES * D1) return;
    float v = out1[t] + b1[t & 63];
    out1[t] = v > 0.0f ? v : expm1f(v);
}

// ---------------------------------------------------------------------------
// Layer 2 (H = 1, C = 2) + pooling
// ---------------------------------------------------------------------------
__global__ void node2_proj(const float* __restrict__ hin, const float* __restrict__ W2,
                           const float* __restrict__ att_s, const float* __restrict__ att_d,
                           float* __restrict__ h2, float* __restrict__ as2,
                           float* __restrict__ ad2, float* __restrict__ emax2) {
    int n = blockIdx.x * blockDim.x + threadIdx.x;
    if (n >= NNODES) return;
    float a0 = 0.f, a1 = 0.f;
    #pragma unroll 8
    for (int j = 0; j < D1; ++j) {
        float v = hin[n * D1 + j];
        a0 += v * W2[j * 2 + 0];
        a1 += v * W2[j * 2 + 1];
    }
    h2[n * 2 + 0] = a0; h2[n * 2 + 1] = a1;
    float s = a0 * att_s[0] + a1 * att_s[1];
    float d = a0 * att_d[0] + a1 * att_d[1];
    as2[n] = s; ad2[n] = d;
    emax2[n] = leaky(s + d);
}

__global__ void edge2_max(const int* __restrict__ src, const int* __restrict__ dst, int E,
                          const float* __restrict__ as2, const float* __restrict__ ad2,
                          float* __restrict__ emax2) {
    int e = blockIdx.x * blockDim.x + threadIdx.x;
    if (e >= E) return;
    atomicMaxF(&emax2[dst[e]], leaky(as2[src[e]] + ad2[dst[e]]));
}

__global__ void node2_denom_init(const float* __restrict__ as2, const float* __restrict__ ad2,
                                 const float* __restrict__ emax2, float* __restrict__ denom2) {
    int n = blockIdx.x * blockDim.x + threadIdx.x;
    if (n >= NNODES) return;
    denom2[n] = __expf(leaky(as2[n] + ad2[n]) - emax2[n]);
}

__global__ void edge2_denom(const int* __restrict__ src, const int* __restrict__ dst, int E,
                            const float* __restrict__ as2, const float* __restrict__ ad2,
                            const float* __restrict__ emax2, float* __restrict__ denom2) {
    int e = blockIdx.x * blockDim.x + threadIdx.x;
    if (e >= E) return;
    int s = src[e], d = dst[e];
    atomicAdd(&denom2[d], __expf(leaky(as2[s] + ad2[d]) - emax2[d]));
}

__global__ void node2_self_msg(const float* __restrict__ h2,
                               const float* __restrict__ as2, const float* __restrict__ ad2,
                               const float* __restrict__ emax2, const float* __restrict__ denom2,
                               float* __restrict__ out2) {
    int n = blockIdx.x * blockDim.x + threadIdx.x;
    if (n >= NNODES) return;
    float es = leaky(as2[n] + ad2[n]);
    float alpha = __expf(es - emax2[n]) / fmaxf(denom2[n], 1e-16f);
    out2[n * 2 + 0] = alpha * h2[n * 2 + 0];
    out2[n * 2 + 1] = alpha * h2[n * 2 + 1];
}

__global__ void edge2_msg(const int* __restrict__ src, const int* __restrict__ dst, int E,
                          const float* __restrict__ h2,
                          const float* __restrict__ as2, const float* __restrict__ ad2,
                          const float* __restrict__ emax2, const float* __restrict__ denom2,
                          float* __restrict__ out2) {
    int e = blockIdx.x * blockDim.x + threadIdx.x;
    if (e >= E) return;
    int s = src[e], d = dst[e];
    float alpha = __expf(leaky(as2[s] + ad2[d]) - emax2[d]) / fmaxf(denom2[d], 1e-16f);
    atomicAdd(&out2[d * 2 + 0], alpha * h2[s * 2 + 0]);
    atomicAdd(&out2[d * 2 + 1], alpha * h2[s * 2 + 1]);
}

__global__ void zero_out(float* __restrict__ out, int n) {
    int t = blockIdx.x * blockDim.x + threadIdx.x;
    if (t < n) out[t] = 0.0f;
}

__global__ void pool(const int* __restrict__ batch, const float* __restrict__ b2,
                     const float* __restrict__ out2, float* __restrict__ out) {
    int t = blockIdx.x * blockDim.x + threadIdx.x;      // t = n*2 + j
    if (t >= NNODES * 2) return;
    int n = t >> 1, j = t & 1;
    atomicAdd(&out[batch[n] * 2 + j], out2[t] + b2[j]);
}

// ---------------------------------------------------------------------------
static inline void* wsTake(char*& p, size_t bytes) {
    void* r = (void*)p;
    p += (bytes + 255) & ~(size_t)255;
    return r;
}

extern "C" void kernel_launch(void* const* d_in, const int* in_sizes, int n_in,
                              void* d_out, int out_size, void* d_ws, size_t ws_size,
                              hipStream_t stream) {
    const float* x    = (const float*)d_in[0];
    const int*   ei   = (const int*)  d_in[1];   // [2, E]
    const int*   bat  = (const int*)  d_in[2];
    const float* W1   = (const float*)d_in[3];
    const float* as1w = (const float*)d_in[4];
    const float* ad1w = (const float*)d_in[5];
    const float* b1   = (const float*)d_in[6];
    const float* W2   = (const float*)d_in[7];
    const float* as2w = (const float*)d_in[8];
    const float* ad2w = (const float*)d_in[9];
    const float* b2   = (const float*)d_in[10];
    float* out = (float*)d_out;

    const int E = in_sizes[1] / 2;
    const int* src = ei;
    const int* dst = ei + E;

    char* w = (char*)d_ws;
    float* h1     = (float*)wsTake(w, (size_t)NNODES * D1 * 4);
    float* out1   = (float*)wsTake(w, (size_t)NNODES * D1 * 4);
    float* a_s1   = (float*)wsTake(w, (size_t)NNODES * 8 * 4);
    float* a_d1   = (float*)wsTake(w, (size_t)NNODES * 8 * 4);
    float* emax1  = (float*)wsTake(w, (size_t)NNODES * 8 * 4);
    float* denom1 = (float*)wsTake(w, (size_t)NNODES * 8 * 4);
    float* h2     = (float*)wsTake(w, (size_t)NNODES * 2 * 4);
    float* out2   = (float*)wsTake(w, (size_t)NNODES * 2 * 4);
    float* a_s2   = (float*)wsTake(w, (size_t)NNODES * 4);
    float* a_d2   = (float*)wsTake(w, (size_t)NNODES * 4);
    float* emax2  = (float*)wsTake(w, (size_t)NNODES * 4);
    float* denom2 = (float*)wsTake(w, (size_t)NNODES * 4);

    const int B = 256;
    auto g = [](long long n, int b) { return (unsigned)((n + b - 1) / b); };

    // ---- Layer 1 ----
    gemm1_wmma<<<g(NNODES, 128), B, 0, stream>>>(x, W1, h1);
    node1_att<<<g((long long)NNODES * 8, B), B, 0, stream>>>(h1, as1w, ad1w, a_s1, a_d1, emax1);
    edge1_max<<<g((long long)E * 8, B), B, 0, stream>>>(src, dst, E, a_s1, a_d1, emax1);
    node1_denom_init<<<g((long long)NNODES * 8, B), B, 0, stream>>>(a_s1, a_d1, emax1, denom1);
    edge1_denom<<<g((long long)E * 8, B), B, 0, stream>>>(src, dst, E, a_s1, a_d1, emax1, denom1);
    node1_self_msg<<<g((long long)NNODES * D1, B), B, 0, stream>>>(h1, a_s1, a_d1, emax1, denom1, out1);
    edge1_msg<<<g((long long)E * 8, B), B, 0, stream>>>(src, dst, E, h1, a_s1, a_d1, emax1, denom1, out1);
    node1_bias_elu<<<g((long long)NNODES * D1, B), B, 0, stream>>>(b1, out1);

    // ---- Layer 2 ----
    node2_proj<<<g(NNODES, B), B, 0, stream>>>(out1, W2, as2w, ad2w, h2, a_s2, a_d2, emax2);
    edge2_max<<<g(E, B), B, 0, stream>>>(src, dst, E, a_s2, a_d2, emax2);
    node2_denom_init<<<g(NNODES, B), B, 0, stream>>>(a_s2, a_d2, emax2, denom2);
    edge2_denom<<<g(E, B), B, 0, stream>>>(src, dst, E, a_s2, a_d2, emax2, denom2);
    node2_self_msg<<<g(NNODES, B), B, 0, stream>>>(h2, a_s2, a_d2, emax2, denom2, out2);
    edge2_msg<<<g(E, B), B, 0, stream>>>(src, dst, E, h2, a_s2, a_d2, emax2, denom2, out2);

    // ---- Pool ----
    zero_out<<<g(out_size, B), B, 0, stream>>>(out, out_size);
    pool<<<g((long long)NNODES * 2, B), B, 0, stream>>>(bat, b2, out2, out);
}
